// _DiffPoolBatchedGraphLayer_5634997092474
// MI455X (gfx1250) — matrix operand order repaired and checked by reading, not compiled, not run on hardware
//
#include <hip/hip_runtime.h>
#include <hip/hip_bf16.h>

// ---------- types for WMMA ----------
typedef __attribute__((ext_vector_type(16))) __bf16 bf16x16;
typedef __attribute__((ext_vector_type(8)))  float  v8f;

union FragAB { uint4 q[2]; bf16x16 v; };

// one k-step worth of operands for a 32x64 wave tile: 2 A-frags, 4 B-frags
struct Frags { FragAB a0, a1, b0, b1, b2, b3; };

__device__ __forceinline__ unsigned short f2bf(float x) {
    unsigned int u = __float_as_uint(x);
    unsigned int r = u + 0x7FFFu + ((u >> 16) & 1u);   // round-to-nearest-even
    return (unsigned short)(r >> 16);
}

__device__ __forceinline__ void load_frags(Frags& f, const unsigned short* pa0,
                                           const unsigned short* pa1,
                                           const unsigned int* pb) {
    f.a0.q[0] = *(const uint4*)(pa0);
    f.a0.q[1] = *(const uint4*)(pa0 + 8);
    f.a1.q[0] = *(const uint4*)(pa1);
    f.a1.q[1] = *(const uint4*)(pa1 + 8);
    f.b0.q[0] = *(const uint4*)(pb);        f.b0.q[1] = *(const uint4*)(pb + 4);
    f.b1.q[0] = *(const uint4*)(pb + 256);  f.b1.q[1] = *(const uint4*)(pb + 260);
    f.b2.q[0] = *(const uint4*)(pb + 512);  f.b2.q[1] = *(const uint4*)(pb + 516);
    f.b3.q[0] = *(const uint4*)(pb + 768);  f.b3.q[1] = *(const uint4*)(pb + 772);
}

#define WMMA_BF16(A, B, C) __builtin_amdgcn_wmma_f32_16x16x32_bf16( \
    false, (A).v, false, (B).v, (short)0, (C), false, false)

__device__ __forceinline__ void mma_frags(const Frags& f, v8f (&acc)[2][4]) {
    acc[0][0] = WMMA_BF16(f.a0, f.b0, acc[0][0]);
    acc[0][1] = WMMA_BF16(f.a0, f.b1, acc[0][1]);
    acc[0][2] = WMMA_BF16(f.a0, f.b2, acc[0][2]);
    acc[0][3] = WMMA_BF16(f.a0, f.b3, acc[0][3]);
    acc[1][0] = WMMA_BF16(f.a1, f.b0, acc[1][0]);
    acc[1][1] = WMMA_BF16(f.a1, f.b1, acc[1][1]);
    acc[1][2] = WMMA_BF16(f.a1, f.b2, acc[1][2]);
    acc[1][3] = WMMA_BF16(f.a1, f.b3, acc[1][3]);
}

// ---------- utility kernels ----------
__global__ void zero_f32(float* __restrict__ p, long n) {
    long i = (long)blockIdx.x * blockDim.x + threadIdx.x;
    long step = (long)gridDim.x * blockDim.x;
    for (; i < n; i += step) p[i] = 0.0f;
}

// dst[512, C] = [Wa(256,C); Wb(256,C)]
__global__ void concat_w(const float* __restrict__ Wa, const float* __restrict__ Wb,
                         float* __restrict__ dst, int C) {
    int tid = blockIdx.x * blockDim.x + threadIdx.x;
    int half = 256 * C;
    if (tid >= 2 * half) return;
    dst[tid] = (tid < half) ? Wa[tid] : Wb[tid - half];
}

// edge scatter: msum[dst] += h[src] (float4 per 4 cols), deg[dst] += 1
__global__ void edge_agg_h(const float* __restrict__ h, const int* __restrict__ src,
                           const int* __restrict__ dst, float* __restrict__ msum,
                           float* __restrict__ deg, long nwork) {
    long tid = (long)blockIdx.x * blockDim.x + threadIdx.x;
    if (tid >= nwork) return;
    long e = tid >> 6;
    int  q = (int)(tid & 63);
    int  s = src[e], d = dst[e];
    const float4 hv = *(const float4*)(h + (long)s * 256 + q * 4);
    float* mp = msum + (long)d * 256 + q * 4;
    atomicAdd(mp + 0, hv.x); atomicAdd(mp + 1, hv.y);
    atomicAdd(mp + 2, hv.z); atomicAdd(mp + 3, hv.w);
    if (q == 0) atomicAdd(deg + d, 1.0f);
}

// X[n, 0:256] = bf16(h[n]); X[n, 256:512] = bf16(msum[n]/max(deg,1))
__global__ void mean_cvt(const float* __restrict__ h, const float* __restrict__ msum,
                         const float* __restrict__ deg, unsigned short* __restrict__ X,
                         long total) {
    long tid = (long)blockIdx.x * blockDim.x + threadIdx.x;
    if (tid >= total) return;
    long n = tid >> 8;
    int  d = (int)(tid & 255);
    float dv = fmaxf(deg[n], 1.0f);
    X[n * 512 + d]       = f2bf(h[tid]);
    X[n * 512 + 256 + d] = f2bf(msum[tid] / dv);
}

// row-major f32 [R, C] -> bf16 B-fragment dwords [kt][nt][lane][v]
__global__ void repack_B(const float* __restrict__ src, unsigned int* __restrict__ dst,
                         int C, long totalDwords) {
    long tid = (long)blockIdx.x * blockDim.x + threadIdx.x;
    if (tid >= totalDwords) return;
    int  v = (int)(tid & 7);
    int  L = (int)((tid >> 3) & 31);
    long f = tid >> 8;
    int  ntiles = C >> 4;
    long kt = f / ntiles;
    int  nt = (int)(f - kt * ntiles);
    int  n  = (nt << 4) + (L & 15);
    long k  = (kt << 5) + ((L >> 4) << 4) + (v << 1);
    float e0 = src[k * C + n];
    float e1 = src[(k + 1) * C + n];
    dst[tid] = (unsigned)f2bf(e0) | ((unsigned)f2bf(e1) << 16);
}

// softmax over rows of 64; writes s (f32, in place) and s^T (bf16, per graph [64,1024])
__global__ __launch_bounds__(256) void softmax_rows(const float* __restrict__ pool,
                                                    float* __restrict__ s,
                                                    unsigned short* __restrict__ sT, int N) {
    int wave = threadIdx.x >> 5, lane = threadIdx.x & 31;
    int row = blockIdx.x * 8 + wave;
    if (row >= N) return;                       // whole-wave uniform exit
    float v0 = pool[(long)row * 64 + lane];
    float v1 = pool[(long)row * 64 + lane + 32];
    float m = fmaxf(v0, v1);
    for (int o = 16; o > 0; o >>= 1) m = fmaxf(m, __shfl_xor(m, o, 32));
    float e0 = __expf(v0 - m), e1 = __expf(v1 - m);
    float sum = e0 + e1;
    for (int o = 16; o > 0; o >>= 1) sum += __shfl_xor(sum, o, 32);
    float inv = 1.0f / sum;
    float s0 = e0 * inv, s1 = e1 * inv;
    s[(long)row * 64 + lane]      = s0;
    s[(long)row * 64 + lane + 32] = s1;
    int g = row >> 10, nl = row & 1023;
    long base = (long)g * 65536 + nl;
    sT[base + (long)lane * 1024]        = f2bf(s0);
    sT[base + (long)(lane + 32) * 1024] = f2bf(s1);
}

// M[dst] += s[src]  (K=64 lanes per edge)
__global__ void edge_agg_s(const float* __restrict__ s, const int* __restrict__ src,
                           const int* __restrict__ dst, float* __restrict__ M, long nwork) {
    long tid = (long)blockIdx.x * blockDim.x + threadIdx.x;
    if (tid >= nwork) return;
    long e = tid >> 6;
    int  l = (int)(tid & 63);
    atomicAdd(&M[(long)dst[e] * 64 + l], s[(long)src[e] * 64 + l]);
}

// ---------- bf16 WMMA GEMM: out = relu?( A(bf16,row-major) @ Bfrag + bias ) ----------
// Each wave computes a 32x64 tile (2 m-tiles x 4 n-tiles), double-buffered over k.
__global__ __launch_bounds__(256) void gemm_bf16_wmma(
    const unsigned short* __restrict__ A, int lda, long a_bs,
    const unsigned int*  __restrict__ Bfrag, int ntiles_total, long b_bs,
    float* __restrict__ out, int ldout, long o_bs,
    const float* __restrict__ bias, int do_relu,
    int kTiles, int mtiles, int ngroups)
{
    int wid  = blockIdx.x * (blockDim.x >> 5) + (threadIdx.x >> 5);
    int lane = threadIdx.x & 31;
    int per_batch = mtiles * ngroups;
    int batch = wid / per_batch;
    int rem   = wid - batch * per_batch;
    int mt    = rem / ngroups;
    int ng    = rem - mt * ngroups;
    int m0    = mt << 5;                 // 32 rows per wave tile
    int nt0   = ng << 2;                 // 4 n-tiles = 64 columns
    int lcol  = lane & 15, lhalf = lane >> 4;

    const unsigned short* pa0 = A + (long)batch * a_bs
                              + (long)(m0 + lcol) * lda + (lhalf ? 16 : 0);
    const unsigned short* pa1 = pa0 + (long)16 * lda;
    const unsigned int*   pb  = Bfrag + (long)batch * b_bs + (nt0 << 8) + (lane << 3);
    const long kstride = (long)ntiles_total << 8;   // dwords per k-step

    v8f acc[2][4] = {{v8f{}, v8f{}, v8f{}, v8f{}}, {v8f{}, v8f{}, v8f{}, v8f{}}};

    Frags f0, f1;
    load_frags(f0, pa0, pa1, pb);
    pa0 += 32; pa1 += 32; pb += kstride;

    for (int i = 0; i < (kTiles >> 1) - 1; ++i) {
        load_frags(f1, pa0, pa1, pb);
        pa0 += 32; pa1 += 32; pb += kstride;
        mma_frags(f0, acc);
        load_frags(f0, pa0, pa1, pb);
        pa0 += 32; pa1 += 32; pb += kstride;
        mma_frags(f1, acc);
    }
    load_frags(f1, pa0, pa1, pb);
    mma_frags(f0, acc);
    mma_frags(f1, acc);

    float* o = out + (long)batch * o_bs;
#pragma unroll
    for (int t = 0; t < 2; ++t) {
#pragma unroll
        for (int j = 0; j < 4; ++j) {
            int n = (nt0 + j) * 16 + lcol;
            float bb = bias ? bias[n] : 0.0f;
#pragma unroll
            for (int r = 0; r < 8; ++r) {
                int m = m0 + t * 16 + r + (lhalf << 3);
                float v = acc[t][j][r] + bb;
                if (do_relu) v = fmaxf(v, 0.0f);
                o[(long)m * ldout + n] = v;
            }
        }
    }
}

// ---------- host launch ----------
extern "C" void kernel_launch(void* const* d_in, const int* in_sizes, int n_in,
                              void* d_out, int out_size, void* d_ws, size_t ws_size,
                              hipStream_t stream) {
    (void)in_sizes; (void)n_in; (void)out_size; (void)ws_size;
    const float* h   = (const float*)d_in[0];
    const int*   src = (const int*)  d_in[1];
    const int*   dst = (const int*)  d_in[2];
    const float* Wsf = (const float*)d_in[3];
    const float* Wnf = (const float*)d_in[4];
    const float* bft = (const float*)d_in[5];
    const float* Wsp = (const float*)d_in[6];
    const float* Wnp = (const float*)d_in[7];
    const float* bpl = (const float*)d_in[8];

    char* ws = (char*)d_ws;
    float*          msum   = (float*)(ws + 0);                    // 33,554,432 B (aliased: feat)
    float*          feat   = msum;
    unsigned short* X      = (unsigned short*)(ws + 33554432ULL); // 33,554,432 B  [N,512] bf16
    float*          deg    = (float*)(ws + 67108864ULL);          // 131,072 B
    float*          WcatF  = (float*)(ws + 67239936ULL);          // 524,288 B
    float*          WcatP  = (float*)(ws + 67764224ULL);          // 131,072 B
    unsigned int*   WfFrag = (unsigned int*)(ws + 67895296ULL);   // 262,144 B
    unsigned int*   WpFrag = (unsigned int*)(ws + 68157440ULL);   // 65,536 B
    float*          pool_s = (float*)(ws + 68222976ULL);          // 8,388,608 B  (pool -> s)
    unsigned short* sT     = (unsigned short*)(ws + 76611584ULL); // 4,194,304 B  per-graph [64,1024]
    float*          Magg   = (float*)(ws + 80805888ULL);          // 8,388,608 B  [N,64]
    unsigned int*   fFrag  = (unsigned int*)(ws + 89194496ULL);   // 16,777,216 B
    unsigned int*   MFrag  = (unsigned int*)(ws + 105971712ULL);  // 4,194,304 B

    float* adj   = (float*)d_out;                  // [2048,2048]
    float* hpool = (float*)d_out + 4194304;        // [2048,256]

    // zero accumulators + block-diagonal output canvas
    zero_f32<<<2048, 256, 0, stream>>>(msum, 8388608L);
    zero_f32<<<64,   256, 0, stream>>>(deg, 32768L);
    zero_f32<<<1024, 256, 0, stream>>>(Magg, 2097152L);
    zero_f32<<<2048, 256, 0, stream>>>(adj, 4194304L);

    // concatenated weight matrices [Wself; Wneigh]
    concat_w<<<512, 256, 0, stream>>>(Wsf, Wnf, WcatF, 256);
    concat_w<<<128, 256, 0, stream>>>(Wsp, Wnp, WcatP, 64);

    // neighbor mean aggregation, then build X = [h | mean] in bf16
    edge_agg_h<<<131072, 256, 0, stream>>>(h, src, dst, msum, deg, 33554432L);
    mean_cvt<<<32768, 256, 0, stream>>>(h, msum, deg, X, 8388608L);

    // weight B-fragments
    repack_B<<<256, 256, 0, stream>>>(WcatF, WfFrag, 256, 65536L);
    repack_B<<<64,  256, 0, stream>>>(WcatP, WpFrag, 64, 16384L);

    // GEMM1: feat = relu(X @ WcatF + b_feat)   [32768,512]@[512,256]
    gemm_bf16_wmma<<<512, 256, 0, stream>>>(X, 512, 0L, WfFrag, 16, 0L,
                                            feat, 256, 0L, bft, 1, 16, 1024, 4);
    // GEMM2: pool = relu(X @ WcatP + b_pool)   [32768,512]@[512,64]
    gemm_bf16_wmma<<<128, 256, 0, stream>>>(X, 512, 0L, WpFrag, 4, 0L,
                                            pool_s, 64, 0L, bpl, 1, 16, 1024, 1);

    // softmax rows -> s (f32) and s^T (bf16 per graph)
    softmax_rows<<<4096, 256, 0, stream>>>(pool_s, pool_s, sT, 32768);

    // M[n] = sum_{e: dst=n} s[src_e]   (replaces dense A: blocks = s^T M)
    edge_agg_s<<<131072, 256, 0, stream>>>(pool_s, src, dst, Magg, 33554432L);

    // B-fragments for pooling GEMMs
    repack_B<<<16384, 256, 0, stream>>>(feat, fFrag, 256, 4194304L);
    repack_B<<<4096,  256, 0, stream>>>(Magg, MFrag, 64, 1048576L);

    // GEMM3: h_pooled[g] = s_g^T @ feat_g   32 x ([64,1024]@[1024,256])
    gemm_bf16_wmma<<<32, 256, 0, stream>>>(sT, 1024, 65536L, fFrag, 16, 131072L,
                                           hpool, 256, 16384L, nullptr, 0, 32, 2, 4);
    // GEMM4: blocks[g] = s_g^T @ M_g written straight into adj block-diagonal
    gemm_bf16_wmma<<<8, 256, 0, stream>>>(sT, 1024, 65536L, MFrag, 4, 32768L,
                                          adj, 2048, 131136L, nullptr, 0, 32, 2, 1);
}
